// DoubleStreamBlock_807453852410
// MI455X (gfx1250) — matrix operand order repaired and checked by reading, not compiled, not run on hardware
//
#include <hip/hip_runtime.h>
#include <cmath>

// ---------- types ----------
typedef __attribute__((ext_vector_type(8)))  float   v8f;
typedef __attribute__((ext_vector_type(4)))  float   v4f;
typedef __attribute__((ext_vector_type(16))) __bf16  v16bf;
typedef __attribute__((ext_vector_type(8)))  __bf16  v8bf;

#define HID   2048
#define MLPD  8192
#define LTXT  256
#define LIMG  2048
#define LSEQ  2304
#define NHEAD 16
#define DH    128

__device__ __forceinline__ v16bf cat8(v8bf a, v8bf b) {
  return __builtin_shufflevector(a, b, 0,1,2,3,4,5,6,7,8,9,10,11,12,13,14,15);
}
__device__ __forceinline__ v8f wmma_bf16(v16bf a, v16bf b, v8f c) {
  // D = A(16x32 bf16) * B(32x16 bf16) + C(16x16 f32)
  return __builtin_amdgcn_wmma_f32_16x16x32_bf16(false, a, false, b, (short)0, c, false, false);
}

// LDS offset (as3 offset = low 32 bits of the generic pointer on amdgcn)
__device__ __forceinline__ unsigned lds_off(const void* p) {
  return (unsigned)(uintptr_t)p;
}
// CDNA5 async DMA: global -> LDS, tracked by ASYNCcnt (no VGPR data path).
// INST_OFFSET is applied to BOTH the LDS and global addresses, so two b128s
// with offset:0/16 copy 32 contiguous bytes.
__device__ __forceinline__ void async_ld16(unsigned lds, unsigned goff, const void* sbase) {
  asm volatile("global_load_async_to_lds_b128 %0, %1, %2"
               :: "v"(lds), "v"(goff), "s"(sbase) : "memory");
}
__device__ __forceinline__ void async_ld32(unsigned lds, unsigned goff, const void* sbase) {
  asm volatile("global_load_async_to_lds_b128 %0, %1, %2\n\t"
               "global_load_async_to_lds_b128 %0, %1, %2 offset:16"
               :: "v"(lds), "v"(goff), "s"(sbase) : "memory");
}
__device__ __forceinline__ void wait_async0() {
#if __has_builtin(__builtin_amdgcn_s_wait_asynccnt)
  __builtin_amdgcn_s_wait_asynccnt(0);
#else
  asm volatile("s_wait_asynccnt 0x0" ::: "memory");
#endif
}

__device__ __forceinline__ float block_sum(float v, float* sbuf) {
  int tid = threadIdx.x;
  sbuf[tid] = v;
  __syncthreads();
  for (int s = blockDim.x >> 1; s > 0; s >>= 1) {
    if (tid < s) sbuf[tid] += sbuf[tid + s];
    __syncthreads();
  }
  float r = sbuf[0];
  __syncthreads();
  return r;
}

// ---------- 1) modulation GEMV, split-K: partial[kseg][n] = silu(vec)[kseg] @ W ----------
__global__ __launch_bounds__(256) void mod_gemv_partial_kernel(
    const float* __restrict__ vec, const float* __restrict__ W,
    float* __restrict__ partial) {
  __shared__ float sv[256];
  const int tid = threadIdx.x;
  const int k0 = blockIdx.y * 256;
  {
    float x = vec[k0 + tid];
    sv[tid] = x / (1.f + __expf(-x));
  }
  __syncthreads();
  const int n = blockIdx.x * 256 + tid;
  float acc = 0.f;
  for (int k = 0; k < 256; ++k) acc = fmaf(sv[k], W[(size_t)(k0 + k) * (6 * HID) + n], acc);
  partial[(size_t)blockIdx.y * (6 * HID) + n] = acc;
}
__global__ __launch_bounds__(256) void mod_reduce_kernel(
    const float* __restrict__ partial, const float* __restrict__ b,
    float* __restrict__ out) {
  const int n = blockIdx.x * 256 + threadIdx.x;
  float s = b[n];
#pragma unroll
  for (int i = 0; i < 8; ++i) s += partial[(size_t)i * (6 * HID) + n];
  out[n] = s;
}

// ---------- 2/7) LayerNorm + modulate: out = (1+sc)*LN(x) + sh  (bf16 out) ----------
__global__ __launch_bounds__(256) void ln_mod_kernel(
    const float* __restrict__ x,
    const float* __restrict__ sh, const float* __restrict__ sc,
    __bf16* __restrict__ out) {
  __shared__ float sbuf[256];
  const int row = blockIdx.x, tid = threadIdx.x;
  const float* xr = x + (size_t)row * HID;
  float vals[8];
  float s = 0.f;
#pragma unroll
  for (int i = 0; i < 8; ++i) { vals[i] = xr[tid + i * 256]; s += vals[i]; }
  const float mu = block_sum(s, sbuf) * (1.f / HID);
  float vs = 0.f;
#pragma unroll
  for (int i = 0; i < 8; ++i) { float d = vals[i] - mu; vs += d * d; }
  const float var = block_sum(vs, sbuf) * (1.f / HID);
  const float inv = rsqrtf(var + 1e-6f);
#pragma unroll
  for (int i = 0; i < 8; ++i) {
    int n = tid + i * 256;
    float y = (vals[i] - mu) * inv;
    out[(size_t)row * HID + n] = (__bf16)((1.f + sc[n]) * y + sh[n]);
  }
}

// ---------- generic GEMM: C[M,N] = A(bf16) @ B(fp32->bf16) [+bias][+epilogue] ----------
// 256 thr (8 waves), tile 64(M) x 128(N), K-step 32, double-buffered LDS.
// A tile staged via async DMA (global_load_async_to_lds_b128, ASYNCcnt).
// B tile prefetched to VGPRs before compute, converted+stored after WMMAs.
// mode 0: outF = acc + bias
// mode 1: outF = resid + gate[n]*(acc + bias)
// mode 2: outB = bf16(gelu_tanh(acc + bias))
__global__ __launch_bounds__(256) void gemm_bf16_kernel(
    const __bf16* __restrict__ A, int lda,
    const float* __restrict__ Bw, int ldb,
    const float* __restrict__ bias,
    float* outF, __bf16* outB, int ldo,
    int K, int mode,
    const float* __restrict__ gate, const float* resid, int ldr) {
  __shared__ __bf16 As[2][64][40];
  __shared__ __bf16 Bs[2][32][136];
  const int tid = threadIdx.x;
  const int lane = tid & 31, wave = tid >> 5;
  const int bm = blockIdx.y * 64, bn = blockIdx.x * 128;
  const int wm = (wave & 3) * 16, wn = (wave >> 2) * 64;
  const int m = lane & 15, kb = (lane >> 4) * 8;
  const int col = lane & 15, hi = lane >> 4;
  const int ar = tid >> 2, ac = (tid & 3) * 8;       // A staging: 16 B / thread
  const int br = tid >> 3, bc = (tid & 7) * 16;      // B staging: 16 f32 / thread

  const unsigned ldsA[2] = { lds_off(&As[0][ar][ac]), lds_off(&As[1][ar][ac]) };
  const unsigned gA0 = (unsigned)(((size_t)(bm + ar) * lda + ac) * sizeof(__bf16));
  const float* gBrow = Bw + (size_t)br * ldb + bn + bc;

  v8f acc[4] = {};

  // prologue: tile 0 into buffer 0
  async_ld16(ldsA[0], gA0, A);
  {
    const float* gp = gBrow;
    v4f f0 = *(const v4f*)(gp + 0), f1 = *(const v4f*)(gp + 4);
    v4f f2 = *(const v4f*)(gp + 8), f3 = *(const v4f*)(gp + 12);
    v8bf h0, h1;
#pragma unroll
    for (int i = 0; i < 4; ++i) {
      h0[i] = (__bf16)f0[i]; h0[4 + i] = (__bf16)f1[i];
      h1[i] = (__bf16)f2[i]; h1[4 + i] = (__bf16)f3[i];
    }
    *(v8bf*)&Bs[0][br][bc]     = h0;
    *(v8bf*)&Bs[0][br][bc + 8] = h1;
  }
  wait_async0();
  __syncthreads();

  const int nsteps = K >> 5;
  for (int it = 0; it < nsteps; ++it) {
    const int cur = it & 1, nxt = cur ^ 1;
    const bool hasn = (it + 1 < nsteps);
    v4f f0, f1, f2, f3;
    if (hasn) {  // prefetch next tile: A via async DMA, B into VGPRs
      const int k1 = (it + 1) << 5;
      async_ld16(ldsA[nxt], gA0 + (unsigned)(k1 * 2), A);
      const float* gp = gBrow + (size_t)k1 * ldb;
      f0 = *(const v4f*)(gp + 0); f1 = *(const v4f*)(gp + 4);
      f2 = *(const v4f*)(gp + 8); f3 = *(const v4f*)(gp + 12);
    }
    // compute on current buffer: batch LDS loads, then 4 back-to-back WMMAs
    const __bf16* ap = &As[cur][wm + m][0];
    v16bf af = cat8(*(const v8bf*)(ap + kb), *(const v8bf*)(ap + 16 + kb));
    v16bf bfr[4];
#pragma unroll
    for (int t = 0; t < 4; ++t) {
      const __bf16* bp = &Bs[cur][0][wn + t * 16 + col];
      for (int s = 0; s < 16; ++s) bfr[t][s] = bp[(s < 8 ? kb + s : 8 + kb + s) * 136];
    }
#pragma unroll
    for (int t = 0; t < 4; ++t) acc[t] = wmma_bf16(af, bfr[t], acc[t]);
    if (hasn) {  // convert + store next B tile after the WMMAs
      v8bf h0, h1;
#pragma unroll
      for (int i = 0; i < 4; ++i) {
        h0[i] = (__bf16)f0[i]; h0[4 + i] = (__bf16)f1[i];
        h1[i] = (__bf16)f2[i]; h1[4 + i] = (__bf16)f3[i];
      }
      *(v8bf*)&Bs[nxt][br][bc]     = h0;
      *(v8bf*)&Bs[nxt][br][bc + 8] = h1;
    }
    wait_async0();
    __syncthreads();
  }

#pragma unroll
  for (int t = 0; t < 4; ++t) {
    const int n = bn + wn + t * 16 + col;
    const float bv = bias[n];
#pragma unroll
    for (int r = 0; r < 8; ++r) {
      const int mm = bm + wm + r + 8 * hi;
      float v = acc[t][r] + bv;
      if (mode == 0) {
        outF[(size_t)mm * ldo + n] = v;
      } else if (mode == 1) {
        outF[(size_t)mm * ldo + n] = resid[(size_t)mm * ldr + n] + gate[n] * v;
      } else {
        float g = 0.5f * v * (1.f + tanhf(0.7978845608028654f * (v + 0.044715f * v * v * v)));
        outB[(size_t)mm * ldo + n] = (__bf16)g;
      }
    }
  }
}

// ---------- 4) per-(l,h): RMSNorm(q,k) + RoPE + scatter to Q[H][L][DH], Kt[H][DH][L], V[H][L][DH] ----------
__global__ __launch_bounds__(128) void rope_qkv_kernel(
    const float* __restrict__ qkv, const float* __restrict__ pe,
    const float* __restrict__ img_qs, const float* __restrict__ img_ks,
    const float* __restrict__ txt_qs, const float* __restrict__ txt_ks,
    __bf16* __restrict__ Q, __bf16* __restrict__ Kt, __bf16* __restrict__ V) {
  __shared__ float sbuf[128];
  __shared__ float qbuf[128];
  __shared__ float kbuf[128];
  const int l = blockIdx.x, h = blockIdx.y, d = threadIdx.x;
  const float* row = qkv + (size_t)l * (3 * HID);
  float q = row[h * DH + d];
  float k = row[HID + h * DH + d];
  float v = row[2 * HID + h * DH + d];
  const bool is_txt = (l < LTXT);
  const float* qs = is_txt ? txt_qs : img_qs;
  const float* ks = is_txt ? txt_ks : img_ks;
  const float sq = block_sum(q * q, sbuf);
  const float sk = block_sum(k * k, sbuf);
  q = q * rsqrtf(sq * (1.f / DH) + 1e-6f) * qs[d];
  k = k * rsqrtf(sk * (1.f / DH) + 1e-6f) * ks[d];
  qbuf[d] = q; kbuf[d] = k;
  __syncthreads();
  const int d2 = d >> 1, j = d & 1;
  const float* p = pe + ((size_t)l * 64 + d2) * 4 + j * 2;  // pe[l, d2, j, 0:2]
  float qr = p[0] * qbuf[2 * d2] + p[1] * qbuf[2 * d2 + 1];
  float kr = p[0] * kbuf[2 * d2] + p[1] * kbuf[2 * d2 + 1];
  qr *= 0.08838834764831845f;  // 1/sqrt(DH) folded into q
  Q [((size_t)h * LSEQ + l) * DH + d]  = (__bf16)qr;
  Kt[((size_t)h * DH + d) * LSEQ + l]  = (__bf16)kr;
  V [((size_t)h * LSEQ + l) * DH + d]  = (__bf16)v;
}

// ---------- 5) flash attention: per wave a 16-query tile, 32-key steps ----------
// K/V tiles double-buffered in LDS via async DMA; all matmuls WMMA bf16.
__global__ __launch_bounds__(256) void attn_kernel(
    const __bf16* __restrict__ Q,   // [H][L][DH]
    const __bf16* __restrict__ Kt,  // [H][DH][L]
    const __bf16* __restrict__ V,   // [H][L][DH]
    __bf16* __restrict__ Aout) {    // [L][H*DH]
  __shared__ __bf16 kt_s[2][128][40];   // K^T tile: rows = d, cols = 32 keys
  __shared__ __bf16 v_s[2][32][136];    // V tile: rows = 32 keys, cols = DH
  __shared__ __bf16 p_s[8][16][40];     // per-wave exp(S) tile 16x32
  const int tid = threadIdx.x, lane = tid & 31, wave = tid >> 5;
  const int h = blockIdx.y;
  const int q0 = blockIdx.x * 128 + wave * 16;
  const int m = lane & 15, kb = (lane >> 4) * 8;
  const int col = lane & 15, hi = lane >> 4;
  const __bf16* Qh  = Q  + (size_t)h * LSEQ * DH;
  const __bf16* Kth = Kt + (size_t)h * DH * LSEQ;
  const __bf16* Vh  = V  + (size_t)h * LSEQ * DH;

  const int dK = tid >> 1, cK = (tid & 1) * 16;   // K^T staging: 32 B / thread
  const int kV = tid >> 3, cV = (tid & 7) * 16;   // V staging: 32 B / thread
  const unsigned ldsK[2] = { lds_off(&kt_s[0][dK][cK]), lds_off(&kt_s[1][dK][cK]) };
  const unsigned ldsV[2] = { lds_off(&v_s[0][kV][cV]),  lds_off(&v_s[1][kV][cV]) };
  const unsigned gK0 = (unsigned)(((size_t)dK * LSEQ + cK) * 2);
  const unsigned gV0 = (unsigned)(((size_t)kV * DH + cV) * 2);

  // prologue: async-stage tile 0 while loading Q fragments
  async_ld32(ldsK[0], gK0, Kth);
  async_ld32(ldsV[0], gV0, Vh);

  v16bf qf[4];
#pragma unroll
  for (int c = 0; c < 4; ++c) {
    const __bf16* qp = Qh + (size_t)(q0 + m) * DH + c * 32;
    qf[c] = cat8(*(const v8bf*)(qp + kb), *(const v8bf*)(qp + 16 + kb));
  }

  float rmax[8], rsum[8], al[8];
  v8f o[8] = {};
#pragma unroll
  for (int r = 0; r < 8; ++r) { rmax[r] = -__builtin_inff(); rsum[r] = 0.f; }

  wait_async0();
  __syncthreads();

  const int nsteps = LSEQ / 32;
  for (int it = 0; it < nsteps; ++it) {
    const int cur = it & 1, nxt = cur ^ 1;
    if (it + 1 < nsteps) {  // async-prefetch next K/V tiles
      const int kt1 = (it + 1) * 32;
      async_ld32(ldsK[nxt], gK0 + (unsigned)(kt1 * 2), Kth);
      async_ld32(ldsV[nxt], gV0 + (unsigned)(kt1 * DH * 2), Vh);
    }

    // scores S(16x32) = Q @ K^T : batch loads, then 4 WMMAs per n-tile
    v8f s0 = {}, s1 = {};
    v16bf bb[4];
#pragma unroll
    for (int c = 0; c < 4; ++c) {
      const __bf16* kp = &kt_s[cur][c * 32][col];
      for (int s = 0; s < 16; ++s) bb[c][s] = kp[(s < 8 ? kb + s : 8 + kb + s) * 40];
    }
#pragma unroll
    for (int c = 0; c < 4; ++c) s0 = wmma_bf16(qf[c], bb[c], s0);
#pragma unroll
    for (int c = 0; c < 4; ++c) {
      const __bf16* kp = &kt_s[cur][c * 32][16 + col];
      for (int s = 0; s < 16; ++s) bb[c][s] = kp[(s < 8 ? kb + s : 8 + kb + s) * 40];
    }
#pragma unroll
    for (int c = 0; c < 4; ++c) s1 = wmma_bf16(qf[c], bb[c], s1);

    // online softmax (rows r+8*hi, cols striped over 16 lanes per half)
#pragma unroll
    for (int r = 0; r < 8; ++r) {
      float mx = fmaxf(s0[r], s1[r]);
      for (int off = 1; off < 16; off <<= 1) mx = fmaxf(mx, __shfl_xor(mx, off, 32));
      float nm = fmaxf(rmax[r], mx);
      float a  = __expf(rmax[r] - nm);
      rmax[r] = nm;
      float p0 = __expf(s0[r] - nm), p1 = __expf(s1[r] - nm);
      s0[r] = p0; s1[r] = p1;
      float ps = p0 + p1;
      for (int off = 1; off < 16; off <<= 1) ps += __shfl_xor(ps, off, 32);
      rsum[r] = rsum[r] * a + ps;
      al[r] = a;
    }

    // transpose P to A-fragment layout via per-wave LDS
#pragma unroll
    for (int r = 0; r < 8; ++r) {
      p_s[wave][r + 8 * hi][col]      = (__bf16)s0[r];
      p_s[wave][r + 8 * hi][16 + col] = (__bf16)s1[r];
    }
    const __bf16* pp = &p_s[wave][m][0];
    v16bf pf = cat8(*(const v8bf*)(pp + kb), *(const v8bf*)(pp + 16 + kb));

    // rescale O, then O += P @ V in groups of 4 n-tiles
#pragma unroll
    for (int t = 0; t < 8; ++t)
#pragma unroll
      for (int r = 0; r < 8; ++r) o[t][r] *= al[r];
#pragma unroll
    for (int g = 0; g < 2; ++g) {
      v16bf vf[4];
#pragma unroll
      for (int t = 0; t < 4; ++t) {
        const __bf16* vp = &v_s[cur][0][(g * 4 + t) * 16 + col];
        for (int s = 0; s < 16; ++s) vf[t][s] = vp[(s < 8 ? kb + s : 8 + kb + s) * 136];
      }
#pragma unroll
      for (int t = 0; t < 4; ++t) o[g * 4 + t] = wmma_bf16(pf, vf[t], o[g * 4 + t]);
    }

    wait_async0();
    __syncthreads();
  }

  // normalize and write attn[l][h*DH + d] (bf16)
#pragma unroll
  for (int t = 0; t < 8; ++t)
#pragma unroll
    for (int r = 0; r < 8; ++r) {
      int rr = q0 + r + 8 * hi;
      int cc = h * DH + t * 16 + col;
      Aout[(size_t)rr * (NHEAD * DH) + cc] = (__bf16)(o[t][r] / rsum[r]);
    }
}

// ---------- host ----------
extern "C" void kernel_launch(void* const* d_in, const int* in_sizes, int n_in,
                              void* d_out, int out_size, void* d_ws, size_t ws_size,
                              hipStream_t stream) {
  (void)in_sizes; (void)n_in; (void)out_size; (void)ws_size;
  const float* img        = (const float*)d_in[0];
  const float* txt        = (const float*)d_in[1];
  const float* vec        = (const float*)d_in[2];
  const float* img_mod_w  = (const float*)d_in[3];
  const float* img_mod_b  = (const float*)d_in[4];
  const float* txt_mod_w  = (const float*)d_in[5];
  const float* txt_mod_b  = (const float*)d_in[6];
  const float* img_qkv_w  = (const float*)d_in[7];
  const float* img_qkv_b  = (const float*)d_in[8];
  const float* img_q_scale= (const float*)d_in[9];
  const float* img_k_scale= (const float*)d_in[10];
  const float* txt_qkv_w  = (const float*)d_in[11];
  const float* txt_qkv_b  = (const float*)d_in[12];
  const float* txt_q_scale= (const float*)d_in[13];
  const float* txt_k_scale= (const float*)d_in[14];
  const float* img_proj_w = (const float*)d_in[15];
  const float* img_proj_b = (const float*)d_in[16];
  const float* txt_proj_w = (const float*)d_in[17];
  const float* txt_proj_b = (const float*)d_in[18];
  const float* img_mlp_w1 = (const float*)d_in[19];
  const float* img_mlp_b1 = (const float*)d_in[20];
  const float* img_mlp_w2 = (const float*)d_in[21];
  const float* img_mlp_b2 = (const float*)d_in[22];
  const float* txt_mlp_w1 = (const float*)d_in[23];
  const float* txt_mlp_b1 = (const float*)d_in[24];
  const float* txt_mlp_w2 = (const float*)d_in[25];
  const float* txt_mlp_b2 = (const float*)d_in[26];
  const float* pe         = (const float*)d_in[27];

  float* out_img = (float*)d_out;                         // [2048][2048]
  float* out_txt = out_img + (size_t)LIMG * HID;          // [256][2048]

  // workspace carve-up (~152 MB)
  float*  mod_img = (float*)d_ws;                         // 12288 f32
  float*  mod_txt = mod_img + 6 * HID;                    // 12288 f32
  __bf16* xnorm   = (__bf16*)(mod_txt + 6 * HID);         // [2304][2048] bf16 (txt rows 0..255, img 256..)
  float*  qkv     = (float*)(xnorm + (size_t)LSEQ * HID); // [2304][6144] f32
  __bf16* Qb      = (__bf16*)(qkv + (size_t)LSEQ * 3 * HID); // [16][2304][128]
  __bf16* Ktb     = Qb  + (size_t)NHEAD * LSEQ * DH;         // [16][128][2304]
  __bf16* Vb      = Ktb + (size_t)NHEAD * LSEQ * DH;         // [16][2304][128]
  __bf16* attn    = Vb  + (size_t)NHEAD * LSEQ * DH;         // [2304][2048]
  __bf16* norm2   = attn + (size_t)LSEQ * HID;               // [2304][2048]
  __bf16* mlph    = norm2 + (size_t)LSEQ * HID;              // [2304][8192]
  float*  modp_img= (float*)(mlph + (size_t)LSEQ * MLPD);    // [8][12288]
  float*  modp_txt= modp_img + 8 * 6 * HID;                  // [8][12288]

  // 1) modulation vectors (split-K, deterministic two-stage reduce)
  mod_gemv_partial_kernel<<<dim3(48, 8), dim3(256), 0, stream>>>(vec, img_mod_w, modp_img);
  mod_gemv_partial_kernel<<<dim3(48, 8), dim3(256), 0, stream>>>(vec, txt_mod_w, modp_txt);
  mod_reduce_kernel<<<dim3(48), dim3(256), 0, stream>>>(modp_img, img_mod_b, mod_img);
  mod_reduce_kernel<<<dim3(48), dim3(256), 0, stream>>>(modp_txt, txt_mod_b, mod_txt);
  // 2) LN + modulate -> bf16
  ln_mod_kernel<<<dim3(LTXT), dim3(256), 0, stream>>>(txt, mod_txt, mod_txt + HID, xnorm);
  ln_mod_kernel<<<dim3(LIMG), dim3(256), 0, stream>>>(img, mod_img, mod_img + HID, xnorm + (size_t)LTXT * HID);
  // 3) QKV GEMMs (fp32 raw out)
  gemm_bf16_kernel<<<dim3(48, 4),  dim3(256), 0, stream>>>(xnorm, HID, txt_qkv_w, 3 * HID, txt_qkv_b,
      qkv, nullptr, 3 * HID, HID, 0, nullptr, nullptr, 0);
  gemm_bf16_kernel<<<dim3(48, 32), dim3(256), 0, stream>>>(xnorm + (size_t)LTXT * HID, HID, img_qkv_w, 3 * HID, img_qkv_b,
      qkv + (size_t)LTXT * 3 * HID, nullptr, 3 * HID, HID, 0, nullptr, nullptr, 0);
  // 4) RMSNorm + RoPE + scatter (K transposed)
  rope_qkv_kernel<<<dim3(LSEQ, NHEAD), dim3(128), 0, stream>>>(qkv, pe,
      img_q_scale, img_k_scale, txt_q_scale, txt_k_scale, Qb, Ktb, Vb);
  // 5) flash attention
  attn_kernel<<<dim3(LSEQ / 128, NHEAD), dim3(256), 0, stream>>>(Qb, Ktb, Vb, attn);
  // 6) output proj + gate + residual -> d_out
  gemm_bf16_kernel<<<dim3(16, 4),  dim3(256), 0, stream>>>(attn, HID, txt_proj_w, HID, txt_proj_b,
      out_txt, nullptr, HID, HID, 1, mod_txt + 2 * HID, txt, HID);
  gemm_bf16_kernel<<<dim3(16, 32), dim3(256), 0, stream>>>(attn + (size_t)LTXT * HID, HID, img_proj_w, HID, img_proj_b,
      out_img, nullptr, HID, HID, 1, mod_img + 2 * HID, img, HID);
  // 7) LN2 + modulate -> bf16
  ln_mod_kernel<<<dim3(LTXT), dim3(256), 0, stream>>>(out_txt, mod_txt + 3 * HID, mod_txt + 4 * HID, norm2);
  ln_mod_kernel<<<dim3(LIMG), dim3(256), 0, stream>>>(out_img, mod_img + 3 * HID, mod_img + 4 * HID,
      norm2 + (size_t)LTXT * HID);
  // 8) MLP1 (+GELU -> bf16)
  gemm_bf16_kernel<<<dim3(64, 4),  dim3(256), 0, stream>>>(norm2, HID, txt_mlp_w1, MLPD, txt_mlp_b1,
      nullptr, mlph, MLPD, HID, 2, nullptr, nullptr, 0);
  gemm_bf16_kernel<<<dim3(64, 32), dim3(256), 0, stream>>>(norm2 + (size_t)LTXT * HID, HID, img_mlp_w1, MLPD, img_mlp_b1,
      nullptr, mlph + (size_t)LTXT * MLPD, MLPD, HID, 2, nullptr, nullptr, 0);
  // 9) MLP2 + gate + residual (in-place on d_out)
  gemm_bf16_kernel<<<dim3(16, 4),  dim3(256), 0, stream>>>(mlph, MLPD, txt_mlp_w2, HID, txt_mlp_b2,
      out_txt, nullptr, HID, MLPD, 1, mod_txt + 5 * HID, out_txt, HID);
  gemm_bf16_kernel<<<dim3(16, 32), dim3(256), 0, stream>>>(mlph + (size_t)LTXT * MLPD, MLPD, img_mlp_w2, HID, img_mlp_b2,
      out_img, nullptr, HID, MLPD, 1, mod_img + 5 * HID, out_img, HID);
}
